// Attention_27659589386599
// MI455X (gfx1250) — matrix operand order repaired and checked by reading, not compiled, not run on hardware
//
#include <hip/hip_runtime.h>

#define BATCH 8
#define SEQ 1024
#define DIMC 1152
#define HEADS 16
#define HD 72
#define HDP 96
#define HDT 112   // HDP + 16 ones-columns for the free row-sum trick

typedef __attribute__((ext_vector_type(16))) __bf16 v16bf;
typedef __attribute__((ext_vector_type(8)))  float  v8f;
typedef __attribute__((ext_vector_type(4)))  unsigned int u32x4;
typedef __attribute__((ext_vector_type(8)))  int i32x8_t;
typedef __attribute__((ext_vector_type(4)))  int i32x4_t;

#if defined(__has_include)
#  if __has_include(<hip/amd_detail/amd_gfx1250_TDM.h>)
#    define TDM_6ARG 1
#  else
#    define TDM_6ARG 0
#  endif
#else
#  define TDM_6ARG 0
#endif

static __device__ __forceinline__ unsigned short f2bf(float f) {
  return __builtin_bit_cast(unsigned short, (__bf16)f);   // native v_cvt path
}
static __device__ __forceinline__ v16bf make_frag(uint4 lo, uint4 hi) {
  union { uint4 q[2]; v16bf v; } u;
  u.q[0] = lo; u.q[1] = hi;
  return u.v;
}

// ---------------------------------------------------------------------------
// Tensor Data Mover: 2D bf16 tile (tile_d0 x tile_d1 elements) -> LDS.
// D# packed per CDNA5 ISA 8.3/8.4 (type=2, count=1, data_size=2B).
// ---------------------------------------------------------------------------
static __device__ __forceinline__ void tdm_load_tile_2d(
    unsigned lds_off, const void* gptr,
    unsigned tile_d0, unsigned tile_d1,
    unsigned tensor_d0, unsigned tensor_d1,
    unsigned long long stride0)
{
  unsigned long long ga = (unsigned long long)(uintptr_t)gptr;
  u32x4 g0 = {0u, 0u, 0u, 0u};
  g0[0] = 1u;                                             // count=1 (valid)
  g0[1] = lds_off;                                        // lds_addr
  g0[2] = (unsigned)ga;                                   // global_addr[31:0]
  g0[3] = (unsigned)((ga >> 32) & 0x01ffffffu) | (2u << 30);  // addr[56:32] | type=2
  i32x8_t g1 = {0, 0, 0, 0, 0, 0, 0, 0};
  g1[0] = (int)(1u << 16);                                // data_size=1 -> 2 bytes
  g1[1] = (int)((tensor_d0 & 0xffffu) << 16);             // tensor_dim0[15:0]
  g1[2] = (int)((tensor_d0 >> 16) | ((tensor_d1 & 0xffffu) << 16));
  g1[3] = (int)((tensor_d1 >> 16) | (tile_d0 << 16));     // tile_dim0
  g1[4] = (int)(tile_d1 & 0xffffu);                       // tile_dim1 (tile_dim2=0)
  g1[5] = (int)(unsigned)(stride0 & 0xffffffffu);         // tensor_dim0_stride lo
  g1[6] = (int)(unsigned)((stride0 >> 32) & 0xffffu);     // stride hi (stride1=0)
  i32x4_t gz = {0, 0, 0, 0};
#if TDM_6ARG
  i32x8_t gz8 = {0, 0, 0, 0, 0, 0, 0, 0};
  __builtin_amdgcn_tensor_load_to_lds(g0, g1, gz, gz, gz8, 0);
#else
  __builtin_amdgcn_tensor_load_to_lds(g0, g1, gz, gz, 0);
#endif
}

// ---------------------------------------------------------------------------
// Tiled bf16 WMMA GEMM: C[M,Nt] = A[M,K] * B[K,Nt] + bias
// EPI==0: A f32 (x) -> scatter to q/k/v bf16 [B,H,N,HDP], q scaled by 1/sqrt(hd)
// EPI==1: A bf16 (attn out) -> f32 out + bias
// 256 threads (8 waves, 2x4 grid); tile 128x128x32; all fragments via b128.
// ---------------------------------------------------------------------------
template <int EPI>
__global__ __launch_bounds__(256) void gemm_qkv_proj(
    const float* __restrict__ Af32, const unsigned short* __restrict__ Abf,
    const float* __restrict__ Bw, const float* __restrict__ bias,
    unsigned short* __restrict__ qout, unsigned short* __restrict__ kout,
    unsigned short* __restrict__ vout, float* __restrict__ fout,
    int M, int K, int Nt)
{
  __shared__ unsigned short As[128][40];    // [m][k]  80B stride: aligned + conflict-free
  __shared__ unsigned short Bs[128][40];    // [n][k]  transposed B

  const int tid    = threadIdx.x;
  const int lane   = tid & 31;
  const int wave   = tid >> 5;
  const int lane16 = lane & 15;
  const int half   = lane >> 4;
  const int wrow   = wave >> 2;
  const int wcol   = wave & 3;
  const int m0     = blockIdx.y * 128;
  const int n0     = blockIdx.x * 128;

  v8f acc[4][2];
  v8f zero8 = {0.f,0.f,0.f,0.f,0.f,0.f,0.f,0.f};
#pragma unroll
  for (int i = 0; i < 4; ++i)
#pragma unroll
    for (int j = 0; j < 2; ++j) acc[i][j] = zero8;

  const int nK = K >> 5;
  for (int kb = 0; kb < nK; ++kb) {
    // ---- A tile (128x32) ----
    if (EPI == 0) {
#pragma unroll
      for (int j = 0; j < 4; ++j) {
        int flat = j * 1024 + tid * 4;
        int row = flat >> 5, col = flat & 31;
        float4 f = *(const float4*)(Af32 + (size_t)(m0 + row) * K + kb * 32 + col);
        uint2 p;
        p.x = (unsigned)f2bf(f.x) | ((unsigned)f2bf(f.y) << 16);
        p.y = (unsigned)f2bf(f.z) | ((unsigned)f2bf(f.w) << 16);
        *(uint2*)&As[row][col] = p;
      }
    } else {
#pragma unroll
      for (int j = 0; j < 4; ++j) {
        int flat = j * 1024 + tid * 4;            // bf16 elems
        int row = flat >> 5, col = flat & 31;
        *(uint2*)&As[row][col] =
            *(const uint2*)(Abf + (size_t)(m0 + row) * K + kb * 32 + col);
      }
    }
    // ---- B tile (32x128) transposed into Bs[n][k] ----
#pragma unroll
    for (int j = 0; j < 4; ++j) {
      int flat = j * 1024 + tid * 4;
      int row = flat >> 7, col = flat & 127;      // row=k, col=n
      float4 f = *(const float4*)(Bw + (size_t)(kb * 32 + row) * Nt + n0 + col);
      Bs[col][row]     = f2bf(f.x);
      Bs[col + 1][row] = f2bf(f.y);
      Bs[col + 2][row] = f2bf(f.z);
      Bs[col + 3][row] = f2bf(f.w);
    }
    __syncthreads();

    // ---- fragments: 2 x b128 each ----
    v16bf af[4];
#pragma unroll
    for (int mt = 0; mt < 4; ++mt) {
      int row = wrow * 64 + mt * 16 + lane16;
      int kb0 = half * 8;
      af[mt] = make_frag(*(const uint4*)&As[row][kb0],
                         *(const uint4*)&As[row][kb0 + 16]);
    }
    v16bf bfr[2];
#pragma unroll
    for (int nt = 0; nt < 2; ++nt) {
      int col = wcol * 32 + nt * 16 + lane16;
      int kbase = half * 8;
      bfr[nt] = make_frag(*(const uint4*)&Bs[col][kbase],
                          *(const uint4*)&Bs[col][kbase + 16]);
    }
#pragma unroll
    for (int mt = 0; mt < 4; ++mt)
#pragma unroll
      for (int nt = 0; nt < 2; ++nt)
        acc[mt][nt] = __builtin_amdgcn_wmma_f32_16x16x32_bf16(
            false, af[mt], false, bfr[nt], (short)0, acc[mt][nt], false, false);
    __syncthreads();
  }

  // ---- epilogue ----
#pragma unroll
  for (int mt = 0; mt < 4; ++mt) {
    const int mbase = m0 + wrow * 64 + mt * 16;
#pragma unroll
    for (int nt = 0; nt < 2; ++nt) {
      const int col = n0 + wcol * 32 + nt * 16 + lane16;
      const float bv = bias[col];
      if (EPI == 0) {
        int which = col / DIMC;
        int rr = col - which * DIMC;
        int head = rr / HD;
        int d = rr - head * HD;
        unsigned short* dst = (which == 0) ? qout : (which == 1) ? kout : vout;
        float scale = (which == 0) ? 0.11785113019775792f : 1.0f;  // 1/sqrt(72)
#pragma unroll
        for (int r = 0; r < 8; ++r) {
          int row = mbase + half * 8 + r;
          int bb = row >> 10, nn = row & (SEQ - 1);
          dst[(((size_t)bb * HEADS + head) * SEQ + nn) * HDP + d] =
              f2bf((acc[mt][nt][r] + bv) * scale);
        }
      } else {
#pragma unroll
        for (int r = 0; r < 8; ++r) {
          int row = mbase + half * 8 + r;
          fout[(size_t)row * Nt + col] = acc[mt][nt][r] + bv;
        }
      }
    }
  }
}

// ---------------------------------------------------------------------------
// Flash attention. Block = (b*h, 128 q rows), 8 waves x 16 rows.
// K tiles streamed by the Tensor Data Mover (double-buffered, wave 0 issues,
// s_wait_tensorcnt + barrier). V stored transposed with a ones-column block
// so rowsum(P) (the softmax denominator) falls out of the 7th P*V WMMA tile.
// ---------------------------------------------------------------------------
__global__ __launch_bounds__(256) void flash_attn(
    const unsigned short* __restrict__ qb, const unsigned short* __restrict__ kb,
    const unsigned short* __restrict__ vb, unsigned short* __restrict__ ob)
{
  __shared__ unsigned short Ks[2][32][HDP];   // TDM destination, double buffered
  __shared__ unsigned short Vs[HDT][40];      // [hd][key] transposed (+ones rows)
  __shared__ unsigned short Ps[8][16][40];    // per-wave P staging (padded)

  const int tid    = threadIdx.x;
  const int lane   = tid & 31;
  const int wave   = tid >> 5;
  const int lane16 = lane & 15;
  const int half   = lane >> 4;
  const int bh     = blockIdx.y;
  const int q0     = blockIdx.x * 128 + wave * 16;
  const int NT     = SEQ / 32;

  // Q fragments straight from global: 2 x global_load_b128 per hd-chunk
  v16bf qf[3];
  {
    const unsigned short* qp = qb + ((size_t)bh * SEQ + q0 + lane16) * HDP;
#pragma unroll
    for (int c = 0; c < 3; ++c) {
      int kb0 = c * 32 + half * 8;
      qf[c] = make_frag(*(const uint4*)(qp + kb0),
                        *(const uint4*)(qp + kb0 + 16));
    }
  }

  // ones-column rows of Vs (hd 96..111): col 96 = 1.0, rest 0 — written once
  for (int i = tid; i < 16 * 40; i += 256) {
    int rrow = 96 + i / 40, cc = i % 40;
    Vs[rrow][cc] = (rrow == 96 && cc < 32) ? (unsigned short)0x3F80 : (unsigned short)0;
  }

  // TDM prefetch of K tile 0
  const unsigned short* kbase_g = kb + (size_t)bh * SEQ * HDP;
  if (wave == 0)
    tdm_load_tile_2d((unsigned)(uintptr_t)&Ks[0][0][0], kbase_g,
                     HDP, 32, HDP, SEQ, HDP);

  float mrow[8];
  v8f zero8 = {0.f,0.f,0.f,0.f,0.f,0.f,0.f,0.f};
  v8f oacc[7];
#pragma unroll
  for (int r = 0; r < 8; ++r) mrow[r] = -1e30f;
#pragma unroll
  for (int t = 0; t < 7; ++t) oacc[t] = zero8;

  const unsigned int* vsrc = (const unsigned int*)(vb + (size_t)bh * SEQ * HDP);

  for (int kt = 0; kt < NT; ++kt) {
    // ---- V tile (32x96) -> Vs transposed [hd][key] ----
    const unsigned int* vs = vsrc + kt * (32 * HDP / 2);
#pragma unroll
    for (int i = 0; i < 6; ++i) {
      int idx = tid + i * 256;          // u32 index in tile
      unsigned int w = vs[idx];
      int key = idx / 48;               // 48 u32 per key row
      int d = (idx - key * 48) * 2;
      Vs[d][key]     = (unsigned short)(w & 0xffffu);
      Vs[d + 1][key] = (unsigned short)(w >> 16);
    }
    // ---- TDM: issue next K tile, wait for current ----
    if (wave == 0) {
      if (kt + 1 < NT) {
        tdm_load_tile_2d((unsigned)(uintptr_t)&Ks[(kt + 1) & 1][0][0],
                         kbase_g + (size_t)(kt + 1) * 32 * HDP,
                         HDP, 32, HDP, SEQ, HDP);
        __builtin_amdgcn_s_wait_tensorcnt(1);   // current tile landed
      } else {
        __builtin_amdgcn_s_wait_tensorcnt(0);
      }
    }
    __syncthreads();

    const unsigned short (*Kb)[HDP] = Ks[kt & 1];

    // ---- S = Q K^T (3 hd-chunks x 2 key-subtiles) ----
    v8f S0 = zero8, S1 = zero8;
#pragma unroll
    for (int c = 0; c < 3; ++c) {
      int kc = c * 32 + half * 8;
      v16bf b0 = make_frag(*(const uint4*)&Kb[lane16][kc],
                           *(const uint4*)&Kb[lane16][kc + 16]);
      v16bf b1 = make_frag(*(const uint4*)&Kb[16 + lane16][kc],
                           *(const uint4*)&Kb[16 + lane16][kc + 16]);
      S0 = __builtin_amdgcn_wmma_f32_16x16x32_bf16(false, qf[c], false, b0,
                                                   (short)0, S0, false, false);
      S1 = __builtin_amdgcn_wmma_f32_16x16x32_bf16(false, qf[c], false, b1,
                                                   (short)0, S1, false, false);
    }

    // ---- online softmax: row-max reduce + rescale; rowsum comes from WMMA ----
#pragma unroll
    for (int r = 0; r < 8; ++r) {
      float s0 = S0[r], s1 = S1[r];
      float rm = fmaxf(s0, s1);
#pragma unroll
      for (int off = 1; off < 16; off <<= 1) rm = fmaxf(rm, __shfl_xor(rm, off, 32));
      float mn = fmaxf(mrow[r], rm);
      float alpha = __expf(mrow[r] - mn);
      mrow[r] = mn;
#pragma unroll
      for (int t = 0; t < 7; ++t) oacc[t][r] *= alpha;
      int prow = half * 8 + r;
      Ps[wave][prow][lane16]      = f2bf(__expf(s0 - mn));
      Ps[wave][prow][16 + lane16] = f2bf(__expf(s1 - mn));
    }
    asm volatile("s_wait_dscnt 0" ::: "memory");  // C-layout -> A-layout staging

    v16bf pf;
    {
      int kb0 = half * 8;
      pf = make_frag(*(const uint4*)&Ps[wave][lane16][kb0],
                     *(const uint4*)&Ps[wave][lane16][kb0 + 16]);
    }

    // ---- O += P V (7 tiles: 6 hd-chunks + ones column => row sums) ----
#pragma unroll
    for (int t = 0; t < 7; ++t) {
      int kbase = half * 8;
      v16bf vf = make_frag(*(const uint4*)&Vs[t * 16 + lane16][kbase],
                           *(const uint4*)&Vs[t * 16 + lane16][kbase + 16]);
      oacc[t] = __builtin_amdgcn_wmma_f32_16x16x32_bf16(false, pf, false, vf,
                                                        (short)0, oacc[t], false, false);
    }
    __syncthreads();
  }

  // ---- finalize: l = oacc[6] col 96 (lane16==0), O /= l, store bf16 ----
  const int bb = bh / HEADS, hh = bh % HEADS;
#pragma unroll
  for (int r = 0; r < 8; ++r) {
    float l = __shfl(oacc[6][r], half * 16, 32);   // broadcast rowsum
    float inv = 1.0f / l;
    int n = q0 + half * 8 + r;
#pragma unroll
    for (int t = 0; t < 6; ++t) {
      int d = t * 16 + lane16;
      if (d < HD)
        ob[((size_t)bb * SEQ + n) * DIMC + hh * HD + d] = f2bf(oacc[t][r] * inv);
    }
  }
}

// ---------------------------------------------------------------------------
extern "C" void kernel_launch(void* const* d_in, const int* in_sizes, int n_in,
                              void* d_out, int out_size, void* d_ws, size_t ws_size,
                              hipStream_t stream) {
  (void)in_sizes; (void)n_in; (void)out_size; (void)ws_size;
  const float* x     = (const float*)d_in[0];
  const float* Wqkv  = (const float*)d_in[1];
  const float* bqkv  = (const float*)d_in[2];
  const float* Wproj = (const float*)d_in[3];
  const float* bproj = (const float*)d_in[4];
  float* out = (float*)d_out;

  const size_t per = (size_t)BATCH * HEADS * SEQ * HDP;
  unsigned short* qb = (unsigned short*)d_ws;
  unsigned short* kb = qb + per;
  unsigned short* vb = kb + per;
  unsigned short* ob = vb + per;                 // [B*N, DIM] bf16

  hipMemsetAsync(qb, 0, 3 * per * sizeof(unsigned short), stream);

  dim3 blk(256);
  gemm_qkv_proj<0><<<dim3(27, 64), blk, 0, stream>>>(
      x, nullptr, Wqkv, bqkv, qb, kb, vb, nullptr, BATCH * SEQ, DIMC, 3 * DIMC);
  flash_attn<<<dim3(SEQ / 128, BATCH * HEADS), blk, 0, stream>>>(qb, kb, vb, ob);
  gemm_qkv_proj<1><<<dim3(9, 64), blk, 0, stream>>>(
      nullptr, ob, Wproj, bproj, nullptr, nullptr, nullptr, out, BATCH * SEQ, DIMC, DIMC);
}
